// HierarchicalScaleNormalization_60447369724332
// MI455X (gfx1250) — matrix operand order repaired and checked by reading, not compiled, not run on hardware
//
#include <hip/hip_runtime.h>
#include <cstdint>

// ---------------------------------------------------------------------------
// MI455X (gfx1250) implementation.
// Compute-bound (~260 GFLOP vs ~300MB @ 23.3 TB/s): all matmul stages run on
// v_wmma_f32_16x16x32_bf16. Every GEMM is TN (A[M,K] row-major, Bt[N,K]
// row-major) so ALL LDS fragment reads are k-contiguous -> ds_load_b128.
// Block tile 128x128, BK=64, 8 waves (2Mx4N), wave tile 64x32 = 16 WMMA/stage.
// ---------------------------------------------------------------------------

typedef __bf16 bf16_t;
typedef __attribute__((ext_vector_type(16))) __bf16 v16bf;
typedef __attribute__((ext_vector_type(8)))  float  v8f;

namespace cfg {
constexpr int B  = 4;
constexpr int S  = 2048;
constexpr int D  = 1024;
constexpr int L  = 2;
constexpr int H  = 8;
constexpr int DH = 128;
constexpr int NTOK = B * S;
}

// ---------------------------------------------------------------------------
// TN GEMM: C[M,N] = alpha * A[M,K] * Bt[N,K]^T
// ---------------------------------------------------------------------------
__global__ __launch_bounds__(256) void gemm_tn_k(
    const bf16_t* __restrict__ A, long long strideA, int lda,
    const bf16_t* __restrict__ Bt, long long strideB, int ldb,
    float* __restrict__ C, long long strideC, int ldc,
    int K, float alpha)
{
  __shared__ bf16_t As[128 * 64];   // [m][k]
  __shared__ bf16_t Bs[128 * 64];   // [n][k]

  const int batch = blockIdx.z;
  A  += (long long)batch * strideA;
  Bt += (long long)batch * strideB;
  C  += (long long)batch * strideC;

  const int t    = threadIdx.x;
  const int lane = t & 31;
  const int wave = t >> 5;
  const int wm   = wave & 1;     // 2 waves along M (64 rows each)
  const int wn   = wave >> 1;    // 4 waves along N (32 cols each)
  const int blockM = blockIdx.y * 128;
  const int blockN = blockIdx.x * 128;

  v8f acc[4][2];
#pragma unroll
  for (int i = 0; i < 4; ++i)
#pragma unroll
    for (int j = 0; j < 2; ++j) acc[i][j] = (v8f){};

  const int sr = t >> 3;          // 0..31
  const int sc = (t & 7) * 8;     // 0,8,...,56

  for (int k0 = 0; k0 < K; k0 += 64) {
    __syncthreads();
    // ---- stage A[128x64] and Bt[128x64] tiles (coalesced b128) ----
#pragma unroll
    for (int i = 0; i < 4; ++i) {
      const int r = sr + 32 * i;
      *(uint4*)&As[r * 64 + sc] =
          *(const uint4*)(A + (long long)(blockM + r) * lda + k0 + sc);
      *(uint4*)&Bs[r * 64 + sc] =
          *(const uint4*)(Bt + (long long)(blockN + r) * ldb + k0 + sc);
    }
    if (k0 + 64 < K) {
      __builtin_prefetch(A + (long long)(blockM + sr) * lda + k0 + 64 + sc, 0, 0);
      __builtin_prefetch(Bt + (long long)(blockN + sr) * ldb + k0 + 64 + sc, 0, 0);
    }
    __syncthreads();

    // ---- two 32-K substeps, 8 WMMAs each ----
#pragma unroll
    for (int s = 0; s < 2; ++s) {
      const int ks = s * 32;
      v16bf a[4], b[2];
      // A fragments: contiguous k order (same as B, not the ISA table order)
      //   lane row = lane&15, K = e + (lane>=16?16:0)
      {
        const int row = lane & 15;
        const int kb  = ks + ((lane >> 4) << 4);
#pragma unroll
        for (int i = 0; i < 4; ++i) {
          const bf16_t* p = &As[(wm * 64 + i * 16 + row) * 64];
#pragma unroll
          for (int e = 0; e < 16; ++e)
            a[i][e] = p[kb + e];
        }
      }
      // B fragments: N = lane&15, K = e + (lane>=16?16:0) -> 16 contiguous
      {
        const int col = lane & 15;
        const int kb  = ks + ((lane >> 4) << 4);
#pragma unroll
        for (int j = 0; j < 2; ++j) {
          const bf16_t* p = &Bs[(wn * 32 + j * 16 + col) * 64 + kb];
#pragma unroll
          for (int e = 0; e < 16; ++e) b[j][e] = p[e];
        }
      }
#pragma unroll
      for (int i = 0; i < 4; ++i)
#pragma unroll
        for (int j = 0; j < 2; ++j)
          acc[i][j] = __builtin_amdgcn_wmma_f32_16x16x32_bf16(
              false, a[i], false, b[j], (short)0, acc[i][j], false, false);
    }
  }

  // ---- epilogue: C/D layout VGPR r -> M = r + (lane>=16?8:0), N = lane&15 ----
  const int colL   = lane & 15;
  const int rowOff = (lane >> 4) << 3;
#pragma unroll
  for (int i = 0; i < 4; ++i) {
#pragma unroll
    for (int j = 0; j < 2; ++j) {
#pragma unroll
      for (int r = 0; r < 8; ++r) {
        const int m0 = blockM + wm * 64 + i * 16 + rowOff + r;
        const int n0 = blockN + wn * 32 + j * 16 + colL;
        C[(long long)m0 * ldc + n0] = alpha * acc[i][j][r];
      }
    }
  }
}

// ---------------------------------------------------------------------------
// Elementwise / transpose helpers
// ---------------------------------------------------------------------------
__global__ __launch_bounds__(256) void cvt_f32_bf16_k(const float* __restrict__ s,
                                                      bf16_t* __restrict__ d,
                                                      long long n) {
  long long i = (long long)blockIdx.x * 256 + threadIdx.x;
  if (i < n) d[i] = (bf16_t)s[i];
}

// Tiled f32 -> bf16 transpose: S[K][N] -> T[N][K]
__global__ __launch_bounds__(256) void cvt_t_k(const float* __restrict__ S,
                                               bf16_t* __restrict__ T,
                                               int K, int N) {
  __shared__ bf16_t tile[32][33];
  const int n0 = blockIdx.x * 32, k0 = blockIdx.y * 32;
  const int tx = threadIdx.x & 31, ty = threadIdx.x >> 5;
#pragma unroll
  for (int i = 0; i < 4; ++i)
    tile[tx][ty + i * 8] = (bf16_t)S[(long long)(k0 + ty + i * 8) * N + n0 + tx];
  __syncthreads();
#pragma unroll
  for (int i = 0; i < 4; ++i)
    T[(long long)(n0 + ty + i * 8) * K + k0 + tx] = tile[ty + i * 8][tx];
}

// C[M,N] f32 + bias -> bf16, transposed: T[N][M]   (used for V -> VT)
__global__ __launch_bounds__(256) void bias_cvt_t_k(const float* __restrict__ C,
                                                    const float* __restrict__ bias,
                                                    bf16_t* __restrict__ T,
                                                    int M, int N) {
  __shared__ bf16_t tile[32][33];
  const int c0 = blockIdx.x * 32, r0 = blockIdx.y * 32;
  const int tx = threadIdx.x & 31, ty = threadIdx.x >> 5;
#pragma unroll
  for (int i = 0; i < 4; ++i) {
    float v = C[(long long)(r0 + ty + i * 8) * N + c0 + tx] + bias[c0 + tx];
    tile[tx][ty + i * 8] = (bf16_t)v;
  }
  __syncthreads();
#pragma unroll
  for (int i = 0; i < 4; ++i)
    T[(long long)(c0 + ty + i * 8) * M + r0 + tx] = tile[ty + i * 8][tx];
}

__global__ __launch_bounds__(256) void zero_u32_k(uint32_t* __restrict__ p, long long n) {
  long long i = (long long)blockIdx.x * 256 + threadIdx.x;
  if (i < n) p[i] = 0u;
}

// C[M,N] (f32, ldc) + bias -> optional SiLU -> bf16 dst (ldd, possibly a slice)
__global__ __launch_bounds__(256) void bias_act_k(const float* __restrict__ C, int ldc,
                                                  const float* __restrict__ bias, int act,
                                                  bf16_t* __restrict__ dst, int ldd,
                                                  int N, long long total) {
  long long i = (long long)blockIdx.x * 256 + threadIdx.x;
  if (i >= total) return;
  long long r = i / N;
  int c = (int)(i % N);
  float v = C[r * ldc + c] + bias[c];
  if (act) v = v / (1.f + __expf(-v));
  dst[r * ldd + c] = (bf16_t)v;
}

// per-token: h = G1_row + p2_b; LN(h)*g+b -> zb[tok][0:D] iff level[tok]==myLevel
__global__ __launch_bounds__(256) void select_ln_k(const float* __restrict__ Hm,
                                                   const float* __restrict__ p2b,
                                                   const float* __restrict__ g,
                                                   const float* __restrict__ bb,
                                                   const int* __restrict__ lev,
                                                   int myLevel,
                                                   bf16_t* __restrict__ zb) {
  const int tok = blockIdx.x;
  if (lev[tok] != myLevel) return;
  const int t = threadIdx.x;
  __shared__ float red[256];
  const float* row = Hm + (long long)tok * cfg::D;
  float v[4];
  float s = 0.f;
#pragma unroll
  for (int i = 0; i < 4; ++i) { int c = t + i * 256; v[i] = row[c] + p2b[c]; s += v[i]; }
  red[t] = s; __syncthreads();
  for (int st = 128; st > 0; st >>= 1) { if (t < st) red[t] += red[t + st]; __syncthreads(); }
  const float mu = red[0] * (1.f / cfg::D); __syncthreads();
  float s2 = 0.f;
#pragma unroll
  for (int i = 0; i < 4; ++i) { float d = v[i] - mu; v[i] = d; s2 += d * d; }
  red[t] = s2; __syncthreads();
  for (int st = 128; st > 0; st >>= 1) { if (t < st) red[t] += red[t + st]; __syncthreads(); }
  const float rstd = rsqrtf(red[0] * (1.f / cfg::D) + 1e-5f);
#pragma unroll
  for (int i = 0; i < 4; ++i) {
    int c = t + i * 256;
    zb[(long long)tok * (2 * cfg::D) + c] = (bf16_t)(v[i] * rstd * g[c] + bb[c]);
  }
}

// row softmax over S=2048; writes bf16 probs and accumulates mean-over-heads
__global__ __launch_bounds__(256) void softmax_k(const float* __restrict__ SC,
                                                 bf16_t* __restrict__ PB,
                                                 float* __restrict__ attn) {
  const long long row = blockIdx.x;  // b*S + q
  const int t = threadIdx.x;
  __shared__ float red[256];
  const float* s = SC + row * cfg::S;
  float v[8];
  float mx = -3.4e38f;
#pragma unroll
  for (int i = 0; i < 8; ++i) { v[i] = s[t + i * 256]; mx = fmaxf(mx, v[i]); }
  red[t] = mx; __syncthreads();
  for (int st = 128; st > 0; st >>= 1) { if (t < st) red[t] = fmaxf(red[t], red[t + st]); __syncthreads(); }
  mx = red[0]; __syncthreads();
  float sum = 0.f;
#pragma unroll
  for (int i = 0; i < 8; ++i) { v[i] = __expf(v[i] - mx); sum += v[i]; }
  red[t] = sum; __syncthreads();
  for (int st = 128; st > 0; st >>= 1) { if (t < st) red[t] += red[t + st]; __syncthreads(); }
  const float inv = 1.f / red[0];
  bf16_t* p = PB + row * cfg::S;
  float* a  = attn + row * cfg::S;
#pragma unroll
  for (int i = 0; i < 8; ++i) {
    int c = t + i * 256;
    float pr = v[i] * inv;
    p[c] = (bf16_t)pr;
    a[c] += pr * (1.f / cfg::H);
  }
}

// final: out = LN(Z_row + i2_b) * g + b  (f32 output)
__global__ __launch_bounds__(256) void ln_out_k(const float* __restrict__ Z,
                                                const float* __restrict__ bias,
                                                const float* __restrict__ g,
                                                const float* __restrict__ bb,
                                                float* __restrict__ out) {
  const int tok = blockIdx.x;
  const int t = threadIdx.x;
  __shared__ float red[256];
  const float* row = Z + (long long)tok * cfg::D;
  float v[4];
  float s = 0.f;
#pragma unroll
  for (int i = 0; i < 4; ++i) { int c = t + i * 256; v[i] = row[c] + bias[c]; s += v[i]; }
  red[t] = s; __syncthreads();
  for (int st = 128; st > 0; st >>= 1) { if (t < st) red[t] += red[t + st]; __syncthreads(); }
  const float mu = red[0] * (1.f / cfg::D); __syncthreads();
  float s2 = 0.f;
#pragma unroll
  for (int i = 0; i < 4; ++i) { float d = v[i] - mu; v[i] = d; s2 += d * d; }
  red[t] = s2; __syncthreads();
  for (int st = 128; st > 0; st >>= 1) { if (t < st) red[t] += red[t + st]; __syncthreads(); }
  const float rstd = rsqrtf(red[0] * (1.f / cfg::D) + 1e-5f);
#pragma unroll
  for (int i = 0; i < 4; ++i) {
    int c = t + i * 256;
    out[(long long)tok * cfg::D + c] = v[i] * rstd * g[c] + bb[c];
  }
}

// ---------------------------------------------------------------------------
// Host-side orchestration
// ---------------------------------------------------------------------------
extern "C" void kernel_launch(void* const* d_in, const int* in_sizes, int n_in,
                              void* d_out, int out_size, void* d_ws, size_t ws_size,
                              hipStream_t stream) {
  using namespace cfg;
  (void)in_sizes; (void)n_in; (void)out_size; (void)ws_size;

  const float* x    = (const float*)d_in[0];
  const int*   lev  = (const int*)d_in[1];
  const float* p1_w = (const float*)d_in[2];
  const float* p1_b = (const float*)d_in[3];
  const float* p2_w = (const float*)d_in[4];
  const float* p2_b = (const float*)d_in[5];
  const float* ln_g = (const float*)d_in[6];
  const float* ln_b = (const float*)d_in[7];
  const float* wq   = (const float*)d_in[8];
  const float* bq   = (const float*)d_in[9];
  const float* wk   = (const float*)d_in[10];
  const float* bk   = (const float*)d_in[11];
  const float* wv   = (const float*)d_in[12];
  const float* bv   = (const float*)d_in[13];
  const float* wo   = (const float*)d_in[14];
  const float* bo   = (const float*)d_in[15];
  const float* i1w  = (const float*)d_in[16];
  const float* i1b  = (const float*)d_in[17];
  const float* i2w  = (const float*)d_in[18];
  const float* i2b  = (const float*)d_in[19];
  const float* ilng = (const float*)d_in[20];
  const float* ilnb = (const float*)d_in[21];

  float* outp = (float*)d_out;
  float* attn = outp + (long long)NTOK * D;   // [B,S,S]

  // ---- workspace carve-out (256B aligned) ----
  char* ws = (char*)d_ws;
  size_t off = 0;
  auto carve = [&](size_t bytes) -> char* {
    char* p = ws + off;
    off = (off + bytes + 255) & ~(size_t)255;
    return p;
  };
  bf16_t* WP1T = (bf16_t*)carve((size_t)L * D * D * 2);   // per level: [D out][D in]^T
  bf16_t* WP2T = (bf16_t*)carve((size_t)L * D * D * 2);
  bf16_t* WQT  = (bf16_t*)carve((size_t)D * D * 2);
  bf16_t* WKT  = (bf16_t*)carve((size_t)D * D * 2);
  bf16_t* WVT  = (bf16_t*)carve((size_t)D * D * 2);
  bf16_t* WOT  = (bf16_t*)carve((size_t)D * D * 2);
  bf16_t* WI1T = (bf16_t*)carve((size_t)2 * D * D * 2);   // [D][2D]
  bf16_t* WI2T = (bf16_t*)carve((size_t)D * D * 2);
  bf16_t* XB   = (bf16_t*)carve((size_t)NTOK * D * 2);
  float*  G1   = (float*) carve((size_t)NTOK * D * 4);
  bf16_t* G1B  = (bf16_t*)carve((size_t)NTOK * D * 2);
  bf16_t* QB   = (bf16_t*)carve((size_t)NTOK * D * 2);
  bf16_t* KB   = (bf16_t*)carve((size_t)NTOK * D * 2);
  bf16_t* VT   = (bf16_t*)carve((size_t)NTOK * D * 2);    // transposed: [D][NTOK]
  bf16_t* ZB   = (bf16_t*)carve((size_t)NTOK * 2 * D * 2);
  float*  SC   = (float*) carve((size_t)B * S * S * 4);
  bf16_t* PB   = (bf16_t*)carve((size_t)B * S * S * 2);
  float*  CTX  = (float*) carve((size_t)NTOK * D * 4);

  auto cvt = [&](const float* s, bf16_t* d, long long n) {
    cvt_f32_bf16_k<<<(unsigned)((n + 255) / 256), 256, 0, stream>>>(s, d, n);
  };
  auto cvt_t = [&](const float* s, bf16_t* d, int K, int N) {
    cvt_t_k<<<dim3((unsigned)(N / 32), (unsigned)(K / 32)), 256, 0, stream>>>(s, d, K, N);
  };
  auto gemm = [&](const bf16_t* A, long long sA, int lda,
                  const bf16_t* Bt, long long sB, int ldb,
                  float* C, long long sC, int ldc,
                  int M, int N, int K, float alpha, int batch) {
    dim3 grid((unsigned)(N / 128), (unsigned)(M / 128), (unsigned)batch);
    gemm_tn_k<<<grid, 256, 0, stream>>>(A, sA, lda, Bt, sB, ldb, C, sC, ldc, K, alpha);
  };
  auto bias_act = [&](const float* C, int ldc, const float* bias, int act,
                      bf16_t* dst, int ldd, int N, long long total) {
    bias_act_k<<<(unsigned)((total + 255) / 256), 256, 0, stream>>>(C, ldc, bias, act, dst, ldd, N, total);
  };

  // ---- weight conversion (transposed to [N][K] bf16) and input conversion ----
  for (int l = 0; l < L; ++l) {
    cvt_t(p1_w + (long long)l * D * D, WP1T + (long long)l * D * D, D, D);
    cvt_t(p2_w + (long long)l * D * D, WP2T + (long long)l * D * D, D, D);
  }
  cvt_t(wq, WQT, D, D);
  cvt_t(wk, WKT, D, D);
  cvt_t(wv, WVT, D, D);
  cvt_t(wo, WOT, D, D);
  cvt_t(i1w, WI1T, 2 * D, D);
  cvt_t(i2w, WI2T, D, D);
  cvt(x, XB, (long long)NTOK * D);

  // ---- zero-init: concat buffer (level>=L tokens stay zero) + attn output ----
  {
    long long nz = (long long)NTOK * D;  // ZB: NTOK*2D bf16 = NTOK*D uint32
    zero_u32_k<<<(unsigned)((nz + 255) / 256), 256, 0, stream>>>((uint32_t*)ZB, nz);
    long long na = (long long)B * S * S;
    zero_u32_k<<<(unsigned)((na + 255) / 256), 256, 0, stream>>>((uint32_t*)attn, na);
  }

  // ---- per-level expert MLPs + LN + routed select into ZB[:, 0:D] ----
  for (int l = 0; l < L; ++l) {
    gemm(XB, 0, D, WP1T + (long long)l * D * D, 0, D, G1, 0, D, NTOK, D, D, 1.f, 1);
    bias_act(G1, D, p1_b + l * D, /*silu*/1, G1B, D, D, (long long)NTOK * D);
    gemm(G1B, 0, D, WP2T + (long long)l * D * D, 0, D, G1, 0, D, NTOK, D, D, 1.f, 1);
    select_ln_k<<<NTOK, 256, 0, stream>>>(G1, p2_b + l * D, ln_g + l * D, ln_b + l * D, lev, l, ZB);
  }

  // ---- QKV projections (A = routed rows in ZB first half, lda = 2D) ----
  gemm(ZB, 0, 2 * D, WQT, 0, D, G1, 0, D, NTOK, D, D, 1.f, 1);
  bias_act(G1, D, bq, 0, QB, D, D, (long long)NTOK * D);
  gemm(ZB, 0, 2 * D, WKT, 0, D, G1, 0, D, NTOK, D, D, 1.f, 1);
  bias_act(G1, D, bk, 0, KB, D, D, (long long)NTOK * D);
  gemm(ZB, 0, 2 * D, WVT, 0, D, G1, 0, D, NTOK, D, D, 1.f, 1);
  bias_cvt_t_k<<<dim3(D / 32, NTOK / 32), 256, 0, stream>>>(G1, bv, VT, NTOK, D);

  // ---- attention: loop heads; scores -> softmax(+mean accum) -> ctx slice ----
  const float isq = 0.08838834764831845f;  // 1/sqrt(128)
  for (int h = 0; h < H; ++h) {
    // scores[b] = (Q_h) (K_h)^T : both token-major, k = head dim -> TN as-is
    gemm(QB + h * DH, (long long)S * D, D,
         KB + h * DH, (long long)S * D, D,
         SC, (long long)S * S, S,
         S, S, DH, isq, B);
    softmax_k<<<NTOK, 256, 0, stream>>>(SC, PB, attn);
    // ctx[b] = P V_h : Bt = VT rows (head dims), k = key index, batch = +b*S
    gemm(PB, (long long)S * S, S,
         VT + (long long)h * DH * NTOK, (long long)S, NTOK,
         CTX + h * DH, (long long)S * D, D,
         S, DH, S, 1.f, B);
  }

  // ---- attended = ctx @ Wo + bo -> ZB[:, D:2D] ----
  cvt(CTX, G1B, (long long)NTOK * D);
  gemm(G1B, 0, D, WOT, 0, D, G1, 0, D, NTOK, D, D, 1.f, 1);
  bias_act(G1, D, bo, 0, ZB + D, 2 * D, D, (long long)NTOK * D);

  // ---- integration MLP: SiLU(z @ i1 + b) @ i2 + b -> LN -> out ----
  gemm(ZB, 0, 2 * D, WI1T, 0, 2 * D, G1, 0, D, NTOK, D, 2 * D, 1.f, 1);
  bias_act(G1, D, i1b, /*silu*/1, G1B, D, D, (long long)NTOK * D);
  gemm(G1B, 0, D, WI2T, 0, D, G1, 0, D, NTOK, D, D, 1.f, 1);
  ln_out_k<<<NTOK, 256, 0, stream>>>(G1, i2b, ilng, ilnb, outp);
}
